// vit_models_73813307949137
// MI455X (gfx1250) — matrix operand order repaired and checked by reading, not compiled, not run on hardware
//
#include <hip/hip_runtime.h>
#include <stdint.h>

// ---------------------------------------------------------------------------
// ViT forward for MI455X (gfx1250): f16 WMMA (16x16x32, f32 accum) for all
// GEMMs + fused flash-attention; f32 master residual/LN/softmax.
// Global->LDS staging uses CDNA5 async-to-LDS copies (ASYNCcnt).
// ---------------------------------------------------------------------------

typedef __attribute__((ext_vector_type(16))) _Float16 v16h;
typedef __attribute__((ext_vector_type(8)))  _Float16 v8h;
typedef __attribute__((ext_vector_type(8)))  float    v8f;

#define EMBED   768
#define NTOK    198
#define BATCH   64
#define NHEAD   12
#define HDIM    64
#define HIDDEN  3072
#define DEPTH   12
#define NPATCH  196
#define KEYPAD  224          // 198 keys padded to 7*32
#define QTILES  13           // ceil(198/16)

// epilogue variants
#define EP_PATCH 0   // + conv bias + pos embed, patch->token row remap, f32 out
#define EP_F16   1   // plain f16 out (qkv)
#define EP_GELU  2   // + bias, exact GELU, f16 out (fc1)
#define EP_RESID 3   // + bias + f32 residual, f32 out (proj, fc2)

// ---- fragment builders for wave32 WMMA operands (callers use k = 16*half + s) -------------
static __device__ __forceinline__ v16h frag2(const _Float16* p0, const _Float16* p1) {
  v8h lo = *(const v8h*)p0;
  v8h hi = *(const v8h*)p1;
  v16h r;
#pragma unroll
  for (int i = 0; i < 8; ++i) { r[i] = lo[i]; r[i + 8] = hi[i]; }
  return r;
}
static __device__ __forceinline__ v16h frag16(const _Float16* p) { return frag2(p, p + 8); }

static __device__ __forceinline__ v8f wmma32(v16h a, v16h b, v8f c) {
  // D = A(16x32 f16) * B(32x16 f16) + C(16x16 f32)
  return __builtin_amdgcn_wmma_f32_16x16x32_f16(false, a, false, b, (short)0, c, false, false);
}

// ---- CDNA5 async global->LDS copy (16 bytes/lane), tracked by ASYNCcnt -----
static __device__ __forceinline__ void async_copy_b128(void* lds_dst, const void* gsrc) {
  uint32_t ldsa = (uint32_t)(uintptr_t)lds_dst;  // LDS aperture: addr[31:0]
  asm volatile("global_load_async_to_lds_b128 %0, %1, off"
               :: "v"(ldsa), "v"(gsrc) : "memory");
}
static __device__ __forceinline__ void async_wait0() {
  asm volatile("s_wait_asynccnt 0x0" ::: "memory");
}

// ---------------------------------------------------------------------------
// f32 -> f16 convert
__global__ void cvt_kernel(const float* __restrict__ in, _Float16* __restrict__ out, long n) {
  size_t i = (size_t)blockIdx.x * blockDim.x + threadIdx.x;
  if (i < (size_t)n) out[i] = (_Float16)in[i];
}

// conv weight (O=768, I*kh*kw=768) row-major -> transposed f16 [K=768][N=768]
__global__ void convt_kernel(const float* __restrict__ w, _Float16* __restrict__ out) {
  size_t i = (size_t)blockIdx.x * blockDim.x + threadIdx.x;
  if (i >= (size_t)EMBED * EMBED) return;
  int c = (int)(i / EMBED);
  int o = (int)(i % EMBED);
  out[(size_t)c * EMBED + o] = (_Float16)w[(size_t)o * EMBED + c];
}

// im2col: x[B,3,224,224] -> patches[B*196, 768] f16 (c = ci*256 + ky*16 + kx)
__global__ void im2col_kernel(const float* __restrict__ x, _Float16* __restrict__ out) {
  size_t i = (size_t)blockIdx.x * blockDim.x + threadIdx.x;
  const size_t total = (size_t)BATCH * NPATCH * EMBED;
  if (i >= total) return;
  int c = (int)(i % EMBED);
  size_t r = i / EMBED;
  int b = (int)(r / NPATCH);
  int p = (int)(r % NPATCH);
  int py = p / 14, px = p % 14;
  int ci = c / 256, rem = c % 256;
  int ky = rem / 16, kx = rem % 16;
  float v = x[(((size_t)b * 3 + ci) * 224 + (py * 16 + ky)) * 224 + (px * 16 + kx)];
  out[i] = (_Float16)v;
}

// local/cls tokens into rows b*198+0 / b*198+1 of the f32 residual stream
__global__ void init_tokens_kernel(const float* __restrict__ loc, const float* __restrict__ cls,
                                   float* __restrict__ h) {
  int b = blockIdx.x;
  for (int c = threadIdx.x; c < EMBED; c += blockDim.x) {
    h[((size_t)b * NTOK + 0) * EMBED + c] = loc[c];
    h[((size_t)b * NTOK + 1) * EMBED + c] = cls[c];
  }
}

// ---------------------------------------------------------------------------
// LayerNorm over 768 (f32 in; f16 out for GEMM input, or f32 out for final)
__global__ __launch_bounds__(256) void ln_kernel(const float* __restrict__ x,
                                                 const float* __restrict__ w,
                                                 const float* __restrict__ bi,
                                                 _Float16* __restrict__ outH,
                                                 float* __restrict__ outF) {
  const int row = blockIdx.x, tid = threadIdx.x;
  const float* xr = x + (size_t)row * EMBED;
  float v[3];
#pragma unroll
  for (int i = 0; i < 3; ++i) v[i] = xr[tid + i * 256];
  __shared__ float red[256];
  red[tid] = v[0] + v[1] + v[2];
  __syncthreads();
  for (int off = 128; off > 0; off >>= 1) {
    if (tid < off) red[tid] += red[tid + off];
    __syncthreads();
  }
  float mu = red[0] * (1.0f / EMBED);
  __syncthreads();
  float d0 = v[0] - mu, d1 = v[1] - mu, d2 = v[2] - mu;
  red[tid] = d0 * d0 + d1 * d1 + d2 * d2;
  __syncthreads();
  for (int off = 128; off > 0; off >>= 1) {
    if (tid < off) red[tid] += red[tid + off];
    __syncthreads();
  }
  float rs = rsqrtf(red[0] * (1.0f / EMBED) + 1e-5f);
#pragma unroll
  for (int i = 0; i < 3; ++i) {
    int c = tid + i * 256;
    float o = (v[i] - mu) * rs * w[c] + bi[c];
    if (outH) outH[(size_t)row * EMBED + c] = (_Float16)o;
    else      outF[(size_t)row * EMBED + c] = o;
  }
}

// ---------------------------------------------------------------------------
// WMMA GEMM: C[M,N] = A[M,K](f16) * Bw[K,N](f16), templated epilogue.
// 128x128 block tile, 256 threads (8 waves of 32x64), K-step 32,
// double-buffered LDS, async global->LDS for A, one barrier per K-step.
#define ASTRIDE 40   // 80B rows: 16B-aligned, conflict-friendly (gcd(20,64)=4)

template <int EP>
__global__ __launch_bounds__(256) void gemm_kernel(
    const _Float16* __restrict__ A, const _Float16* __restrict__ Bw,
    int M, int N, int K,
    const float* __restrict__ bias, const float* __restrict__ resid,
    const float* __restrict__ pos,
    float* __restrict__ outF, _Float16* __restrict__ outH) {
  const int tid = threadIdx.x;
  const int lane = tid & 31;
  const int wv = tid >> 5;
  const int waveRow = (wv & 3) * 32;   // 4 wave-rows x 2 wave-cols
  const int waveCol = (wv >> 2) * 64;
  const long blockM = (long)blockIdx.y * 128;
  const long blockN = (long)blockIdx.x * 128;

  __shared__ _Float16 sA[2][128 * ASTRIDE];   // [row][k]
  __shared__ _Float16 sB[2][128 * ASTRIDE];   // transposed: [n][kk]

  v8f cc[2][4];
#pragma unroll
  for (int mt = 0; mt < 2; ++mt)
#pragma unroll
    for (int nt = 0; nt < 4; ++nt)
#pragma unroll
      for (int j = 0; j < 8; ++j) cc[mt][nt][j] = 0.0f;

  const int arow  = tid >> 1;            // 0..127
  const int akoff = (tid & 1) * 16;      // 0 / 16
  const int bkk   = tid >> 3;            // 0..31
  const int bnoff = (tid & 7) * 16;      // 0..112

  const int frow  = lane & 15;
  const int abase = (lane < 16) ? 0 : 8;   // unused; A-frag K base is bko
  const int bko   = (lane < 16) ? 0 : 16;  // B-frag K base

  auto stageA = [&](int k0, int buf) {   // 2 async b128 copies / thread
    const _Float16* g = A + (size_t)(blockM + arow) * K + k0 + akoff;
    _Float16* d = &sA[buf][arow * ASTRIDE + akoff];
    async_copy_b128(d, g);
    async_copy_b128(d + 8, g + 8);
  };
  auto loadB = [&](int k0, v8h& b0, v8h& b1) {
    const _Float16* g = Bw + (size_t)(k0 + bkk) * N + blockN + bnoff;
    b0 = *(const v8h*)g;
    b1 = *(const v8h*)(g + 8);
  };
  auto storeB = [&](int buf, v8h b0, v8h b1) {   // transpose into [n][kk]
#pragma unroll
    for (int i = 0; i < 8; ++i) {
      sB[buf][(bnoff + i) * ASTRIDE + bkk]     = b0[i];
      sB[buf][(bnoff + 8 + i) * ASTRIDE + bkk] = b1[i];
    }
  };

  const int nk = K / 32;
  {
    stageA(0, 0);
    v8h b0, b1;
    loadB(0, b0, b1);
    storeB(0, b0, b1);
    async_wait0();
  }
  __syncthreads();

  for (int i = 0; i < nk; ++i) {
    const int cur = i & 1, nxt = cur ^ 1;
    const bool pre = (i + 1 < nk);
    v8h nb0, nb1;
    if (pre) {                       // copies/loads in flight during WMMAs
      stageA((i + 1) * 32, nxt);
      loadB((i + 1) * 32, nb0, nb1);
    }
    v16h aF[2];
#pragma unroll
    for (int mt = 0; mt < 2; ++mt) {
      const _Float16* p = &sA[cur][(waveRow + mt * 16 + frow) * ASTRIDE];
      aF[mt] = frag16(p + bko);
    }
#pragma unroll
    for (int nt = 0; nt < 4; ++nt) {
      v16h bF = frag16(&sB[cur][(waveCol + nt * 16 + frow) * ASTRIDE + bko]);
#pragma unroll
      for (int mt = 0; mt < 2; ++mt) cc[mt][nt] = wmma32(aF[mt], bF, cc[mt][nt]);
    }
    if (pre) {
      storeB(nxt, nb0, nb1);
      async_wait0();
    }
    __syncthreads();
  }

  // epilogue (compile-time specialized)
  const int mrow = (lane < 16) ? 0 : 8;
#pragma unroll
  for (int mt = 0; mt < 2; ++mt)
#pragma unroll
    for (int nt = 0; nt < 4; ++nt)
#pragma unroll
      for (int j = 0; j < 8; ++j) {
        long gr = blockM + waveRow + mt * 16 + j + mrow;
        int  gc = (int)(blockN + waveCol + nt * 16 + (lane & 15));
        float v = cc[mt][nt][j];
        if constexpr (EP == EP_PATCH) {
          long bb = gr / NPATCH;
          long p  = gr - bb * NPATCH;
          long orow = bb * NTOK + 2 + p;       // tokens 2..197
          v += bias[gc] + pos[p * EMBED + gc];
          outF[(size_t)orow * N + gc] = v;
        } else if constexpr (EP == EP_F16) {
          outH[(size_t)gr * N + gc] = (_Float16)v;
        } else if constexpr (EP == EP_GELU) {
          v += bias[gc];
          v = 0.5f * v * (1.0f + erff(v * 0.70710678118f));
          outH[(size_t)gr * N + gc] = (_Float16)v;
        } else {  // EP_RESID
          v += bias[gc] + resid[(size_t)gr * N + gc];
          outF[(size_t)gr * N + gc] = v;
        }
      }
}

// ---------------------------------------------------------------------------
// Fused flash attention. Block = 64 threads (2 waves) per (b,h).
// K staged via async-to-LDS ([224][64] row-major); V transposed manually
// ([64][224]). Each wave owns q-tiles qt = wv, wv+2, ... (13 tiles).
__global__ __launch_bounds__(64) void attn_kernel(const _Float16* __restrict__ qkv,
                                                  _Float16* __restrict__ out) {
  const int tid  = threadIdx.x;
  const int lane = tid & 31;
  const int wv   = tid >> 5;
  const int h = blockIdx.x % NHEAD;
  const int b = blockIdx.x / NHEAD;

  __shared__ _Float16 sK[KEYPAD * HDIM];   // [key][d]
  __shared__ _Float16 sVT[HDIM * KEYPAD];  // [d][key]
  __shared__ float    sS[2][16 * 32];
  __shared__ _Float16 sP[2][16 * 32];
  __shared__ float    sM[2][16], sL[2][16], sFac[2][16];

  const size_t tokBase = (size_t)b * NTOK;

  // K: async global->LDS copies (ASYNCcnt)
  for (int idx = tid; idx < NTOK * 8; idx += 64) {
    int r = idx >> 3, ch = idx & 7;
    async_copy_b128(&sK[r * HDIM + ch * 8],
                    qkv + (tokBase + r) * 2304 + EMBED + h * HDIM + ch * 8);
  }
  // zero-pad keys 198..223
  for (int idx = tid; idx < (KEYPAD - NTOK) * 8; idx += 64) {
    int r = NTOK + (idx >> 3), ch = idx & 7;
    v8h z;
#pragma unroll
    for (int i = 0; i < 8; ++i) z[i] = (_Float16)0.0f;
    *(v8h*)&sK[r * HDIM + ch * 8] = z;
  }
  // V: manual transposed store
  for (int idx = tid; idx < KEYPAD * 8; idx += 64) {
    int r = idx >> 3, ch = idx & 7;
    v8h vv;
    if (r < NTOK) {
      vv = *(const v8h*)(qkv + (tokBase + r) * 2304 + 2 * EMBED + h * HDIM + ch * 8);
    } else {
#pragma unroll
      for (int i = 0; i < 8; ++i) vv[i] = (_Float16)0.0f;
    }
#pragma unroll
    for (int i = 0; i < 8; ++i) sVT[(ch * 8 + i) * KEYPAD + r] = vv[i];
  }
  async_wait0();
  __syncthreads();   // only block-wide barrier; waves independent after this

  float*    mySS = sS[wv];
  _Float16* myP  = sP[wv];
  float*    myM  = sM[wv];
  float*    myL  = sL[wv];
  float*    myF  = sFac[wv];

  const int frow  = lane & 15;
  const int abase = (lane < 16) ? 0 : 8;
  const int bko   = (lane < 16) ? 0 : 16;
  const int mrow  = (lane < 16) ? 0 : 8;
  const int bn    = lane & 15;

  for (int qt = wv; qt < QTILES; qt += 2) {
    const int row0 = (qt < QTILES - 1) ? qt * 16 : (NTOK - 16);  // overlap last

    const _Float16* qp = qkv + (tokBase + row0 + frow) * 2304 + h * HDIM;
    v16h qf0 = frag16(qp + bko);
    v16h qf1 = frag16(qp + 32 + bko);

    v8f o[4];
#pragma unroll
    for (int g = 0; g < 4; ++g)
#pragma unroll
      for (int j = 0; j < 8; ++j) o[g][j] = 0.0f;
    if (lane < 16) { myM[lane] = -1e30f; myL[lane] = 0.0f; }

    for (int kt = 0; kt < KEYPAD; kt += 32) {
      v8f s0, s1;
#pragma unroll
      for (int j = 0; j < 8; ++j) { s0[j] = 0.0f; s1[j] = 0.0f; }
      {
        v16h bk = frag16(&sK[(kt + bn) * HDIM + bko]);
        s0 = wmma32(qf0, bk, s0);
        bk = frag16(&sK[(kt + bn) * HDIM + 32 + bko]);
        s0 = wmma32(qf1, bk, s0);
        bk = frag16(&sK[(kt + 16 + bn) * HDIM + bko]);
        s1 = wmma32(qf0, bk, s1);
        bk = frag16(&sK[(kt + 16 + bn) * HDIM + 32 + bko]);
        s1 = wmma32(qf1, bk, s1);
      }
#pragma unroll
      for (int j = 0; j < 8; ++j) {
        int r = j + mrow;
        mySS[r * 32 + bn]      = (kt + bn < NTOK)      ? s0[j] * 0.125f : -1e30f;
        mySS[r * 32 + 16 + bn] = (kt + 16 + bn < NTOK) ? s1[j] * 0.125f : -1e30f;
      }
      if (lane < 16) {   // online softmax, one q-row per lane
        int r = lane;
        float m = myM[r], tmax = -1e30f;
#pragma unroll
        for (int c2 = 0; c2 < 32; ++c2) tmax = fmaxf(tmax, mySS[r * 32 + c2]);
        float mn  = fmaxf(m, tmax);
        float fac = expf(m - mn);
        float l   = myL[r] * fac;
#pragma unroll
        for (int c2 = 0; c2 < 32; ++c2) {
          float p = expf(mySS[r * 32 + c2] - mn);
          l += p;
          myP[r * 32 + c2] = (_Float16)p;
        }
        myM[r] = mn; myL[r] = l; myF[r] = fac;
      }
      float fr[8];
#pragma unroll
      for (int j = 0; j < 8; ++j) fr[j] = myF[j + mrow];
#pragma unroll
      for (int g = 0; g < 4; ++g)
#pragma unroll
        for (int j = 0; j < 8; ++j) o[g][j] *= fr[j];
      v16h ap = frag16(&myP[frow * 32 + bko]);
#pragma unroll
      for (int g = 0; g < 4; ++g) {
        v16h bv = frag16(&sVT[(g * 16 + bn) * KEYPAD + kt + bko]);
        o[g] = wmma32(ap, bv, o[g]);
      }
    }

    float li[8];
#pragma unroll
    for (int j = 0; j < 8; ++j) li[j] = 1.0f / myL[j + mrow];
#pragma unroll
    for (int j = 0; j < 8; ++j) {
      size_t token = tokBase + row0 + j + mrow;
#pragma unroll
      for (int g = 0; g < 4; ++g)
        out[token * EMBED + h * HDIM + g * 16 + bn] = (_Float16)(o[g][j] * li[j]);
    }
  }
}

// ---------------------------------------------------------------------------
extern "C" void kernel_launch(void* const* d_in, const int* in_sizes, int n_in,
                              void* d_out, int out_size, void* d_ws, size_t ws_size,
                              hipStream_t stream) {
  (void)in_sizes; (void)n_in; (void)out_size; (void)ws_size;
  const float* x      = (const float*)d_in[0];
  const float* conv_w = (const float*)d_in[1];
  const float* conv_b = (const float*)d_in[2];
  const float* pos    = (const float*)d_in[3];
  const float* cls    = (const float*)d_in[4];
  const float* loc    = (const float*)d_in[5];
  const float* ln1w   = (const float*)d_in[6];
  const float* ln1b   = (const float*)d_in[7];
  const float* qkvw   = (const float*)d_in[8];
  const float* projw  = (const float*)d_in[9];
  const float* projb  = (const float*)d_in[10];
  const float* ln2w   = (const float*)d_in[11];
  const float* ln2b   = (const float*)d_in[12];
  const float* fc1w   = (const float*)d_in[13];
  const float* fc1b   = (const float*)d_in[14];
  const float* fc2w   = (const float*)d_in[15];
  const float* fc2b   = (const float*)d_in[16];
  const float* lnfw   = (const float*)d_in[17];
  const float* lnfb   = (const float*)d_in[18];

  const size_t ROWS = (size_t)BATCH * NTOK;    // 12672
  const size_t PR   = (size_t)BATCH * NPATCH;  // 12544

  char* ws = (char*)d_ws;
  size_t off = 0;
  auto carve = [&](size_t bytes) -> char* {
    char* p = ws + off;
    off += (bytes + 255) & ~(size_t)255;
    return p;
  };
  float*     hbuf    = (float*)    carve(ROWS * EMBED * 4);
  _Float16*  ybuf    = (_Float16*) carve(ROWS * EMBED * 2);
  _Float16*  qkvbuf  = (_Float16*) carve(ROWS * 3 * EMBED * 2);
  _Float16*  attnbuf = (_Float16*) carve(ROWS * EMBED * 2);
  _Float16*  ffnbuf  = (_Float16*) carve(ROWS * HIDDEN * 2);
  _Float16*  patches = (_Float16*) carve(PR * EMBED * 2);
  _Float16*  wqkv    = (_Float16*) carve((size_t)DEPTH * EMBED * 3 * EMBED * 2);
  _Float16*  wproj   = (_Float16*) carve((size_t)DEPTH * EMBED * EMBED * 2);
  _Float16*  wfc1    = (_Float16*) carve((size_t)DEPTH * EMBED * HIDDEN * 2);
  _Float16*  wfc2    = (_Float16*) carve((size_t)DEPTH * HIDDEN * EMBED * 2);
  _Float16*  wconvT  = (_Float16*) carve((size_t)EMBED * EMBED * 2);

  auto cvt = [&](const float* src, _Float16* dst, long n) {
    cvt_kernel<<<(unsigned)((n + 255) / 256), 256, 0, stream>>>(src, dst, n);
  };
  cvt(qkvw,  wqkv,  (long)DEPTH * EMBED * 3 * EMBED);
  cvt(projw, wproj, (long)DEPTH * EMBED * EMBED);
  cvt(fc1w,  wfc1,  (long)DEPTH * EMBED * HIDDEN);
  cvt(fc2w,  wfc2,  (long)DEPTH * HIDDEN * EMBED);
  convt_kernel<<<(EMBED * EMBED + 255) / 256, 256, 0, stream>>>(conv_w, wconvT);

  im2col_kernel<<<(unsigned)((PR * EMBED + 255) / 256), 256, 0, stream>>>(x, patches);

  // patch embed GEMM: [12544,768] x [768,768] (+conv_b, +pos, rowmap -> hbuf)
  gemm_kernel<EP_PATCH><<<dim3(EMBED / 128, (unsigned)(PR / 128)), 256, 0, stream>>>(
      patches, wconvT, (int)PR, EMBED, EMBED, conv_b, nullptr, pos, hbuf, nullptr);
  init_tokens_kernel<<<BATCH, 256, 0, stream>>>(loc, cls, hbuf);

  for (int l = 0; l < DEPTH; ++l) {
    ln_kernel<<<(unsigned)ROWS, 256, 0, stream>>>(
        hbuf, ln1w + (size_t)l * EMBED, ln1b + (size_t)l * EMBED, ybuf, nullptr);
    gemm_kernel<EP_F16><<<dim3(3 * EMBED / 128, (unsigned)(ROWS / 128)), 256, 0, stream>>>(
        ybuf, wqkv + (size_t)l * EMBED * 3 * EMBED, (int)ROWS, 3 * EMBED, EMBED,
        nullptr, nullptr, nullptr, nullptr, qkvbuf);
    attn_kernel<<<BATCH * NHEAD, 64, 0, stream>>>(qkvbuf, attnbuf);
    gemm_kernel<EP_RESID><<<dim3(EMBED / 128, (unsigned)(ROWS / 128)), 256, 0, stream>>>(
        attnbuf, wproj + (size_t)l * EMBED * EMBED, (int)ROWS, EMBED, EMBED,
        projb + (size_t)l * EMBED, hbuf, nullptr, hbuf, nullptr);
    ln_kernel<<<(unsigned)ROWS, 256, 0, stream>>>(
        hbuf, ln2w + (size_t)l * EMBED, ln2b + (size_t)l * EMBED, ybuf, nullptr);
    gemm_kernel<EP_GELU><<<dim3(HIDDEN / 128, (unsigned)(ROWS / 128)), 256, 0, stream>>>(
        ybuf, wfc1 + (size_t)l * EMBED * HIDDEN, (int)ROWS, HIDDEN, EMBED,
        fc1b + (size_t)l * HIDDEN, nullptr, nullptr, nullptr, ffnbuf);
    gemm_kernel<EP_RESID><<<dim3(EMBED / 128, (unsigned)(ROWS / 128)), 256, 0, stream>>>(
        ffnbuf, wfc2 + (size_t)l * HIDDEN * EMBED, (int)ROWS, EMBED, HIDDEN,
        fc2b + (size_t)l * EMBED, hbuf, nullptr, hbuf, nullptr);
  }

  ln_kernel<<<(unsigned)ROWS, 256, 0, stream>>>(hbuf, lnfw, lnfb, nullptr, (float*)d_out);
}
